// GATE_10685878633221
// MI455X (gfx1250) — matrix operand order repaired and checked by reading, not compiled, not run on hardware
//
#include <hip/hip_runtime.h>

// ---------------------------------------------------------------------------
// GATv2 GNN forward for MI455X (gfx1250, wave32).
// Dense transforms: bf16 WMMA GEMM (v_wmma_f32_16x16x32_bf16, fp32 acc).
// Block tile 128x64, 8 waves as 4x2 grid of 32x32 wave tiles -> 4 WMMA per
// wave per K-step. Staging uses GLOBAL_LOAD_ASYNC_TO_LDS_B128 (ASYNCcnt) so
// global->LDS traffic bypasses VGPRs and overlaps with WMMA.
// Edge attention fused: logit -> segment max -> exp/sum -> weighted scatter.
// eu2/ebn2 are dead wrt the output and skipped.
// ---------------------------------------------------------------------------

typedef __attribute__((ext_vector_type(16))) __bf16 v16bf;
typedef __attribute__((ext_vector_type(8)))  __bf16 v8bf;
typedef __attribute__((ext_vector_type(8)))  float  v8f;

#define N_NODES  20000
#define N_EDGES  160000
#define N_GRAPHS 64

// ---------------- helpers ----------------

__device__ __forceinline__ unsigned short f2bf(float f) {
  unsigned int u = __float_as_uint(f);
  unsigned int r = u + 0x7fffu + ((u >> 16) & 1u);   // round-to-nearest-even
  return (unsigned short)(r >> 16);
}

__device__ __forceinline__ int f2ord(float f) {      // order-preserving f32->i32
  int i = __float_as_int(f);
  return i >= 0 ? i : (i ^ 0x7fffffff);
}
__device__ __forceinline__ float ord2f(int i) {
  return __int_as_float(i >= 0 ? i : (i ^ 0x7fffffff));
}

// async global -> LDS copy of 16 bytes (GV mode, SADDR = off).
// lds_addr: LDS byte address (low 32 bits of the generic shared pointer);
// data lands in LDS directly, tracked by ASYNCcnt.
__device__ __forceinline__ void async_ld_b128(unsigned lds_addr, const void* gptr) {
  asm volatile("global_load_async_to_lds_b128 %0, %1, off"
               :: "v"(lds_addr), "v"(gptr) : "memory");
}
__device__ __forceinline__ void wait_asynccnt0() {
  asm volatile("s_wait_asynccnt 0x0" ::: "memory");
}

// ---------------- conversion kernels ----------------

// A operand: f32 [M,K] -> bf16 [M,Kp], zero-padding K..Kp-1
__global__ void cvt_pad_bf16(const float* __restrict__ in,
                             unsigned short* __restrict__ out,
                             long M, int K, int Kp) {
  long i = (long)blockIdx.x * blockDim.x + threadIdx.x;
  if (i >= M * Kp) return;
  long m = i / Kp;
  int  k = (int)(i - m * Kp);
  out[i] = (k < K) ? f2bf(in[m * K + k]) : (unsigned short)0;
}

// B operand: f32 [K,N] -> bf16 B^T [N,Kp], zero-padding K..Kp-1
__global__ void cvtT_bf16(const float* __restrict__ in,
                          unsigned short* __restrict__ out,
                          int K, int N, int Kp) {
  long i = (long)blockIdx.x * blockDim.x + threadIdx.x;
  if (i >= (long)N * Kp) return;
  int n = (int)(i / Kp);
  int k = (int)(i - (long)n * Kp);
  out[i] = (k < K) ? f2bf(in[(long)k * N + n]) : (unsigned short)0;
}

// ---------------- fill / elementwise kernels ----------------

__global__ void fill_f32(float* p, float v, long n) {
  long i = (long)blockIdx.x * blockDim.x + threadIdx.x;
  if (i < n) p[i] = v;
}
__global__ void fill_i32(int* p, int v, long n) {
  long i = (long)blockIdx.x * blockDim.x + threadIdx.x;
  if (i < n) p[i] = v;
}

__global__ void bias_relu_inplace(float* __restrict__ X,
                                  const float* __restrict__ bias, long M, int N) {
  long i = (long)blockIdx.x * blockDim.x + threadIdx.x;
  if (i < M * (long)N) {
    int c = (int)(i % N);
    X[i] = fmaxf(X[i] + bias[c], 0.f);
  }
}

// ---------------- bf16 WMMA GEMM ----------------
// C[M,N] = act(A @ B + bias). A: bf16 [Mp,Kp] (Mp = ceil128(M), pad rows
// garbage -> per-row independence keeps valid rows exact; epilogue guards).
// B^T: bf16 [N,Kp]. Requires N % 64 == 0, Kp % 32 == 0.
// 256 threads = 8 waves (4 m x 2 n); wave tile 32x32 = 4 WMMAs per K-step.

__global__ __launch_bounds__(256)
void gemm_bf16_wmma(const unsigned short* __restrict__ A,
                    const unsigned short* __restrict__ BT,
                    const float* __restrict__ bias,
                    float* __restrict__ C,
                    int M, int N, int Kp, int relu) {
  // 48-ushort stride = 96B rows: fragment reads stay 16B/32B aligned
  __shared__ __align__(32) unsigned short As[128][48];   // [m][k]  12 KB
  __shared__ __align__(32) unsigned short Bs[64][48];    // [n][k]   6 KB

  const int row0 = blockIdx.y * 128;
  const int col0 = blockIdx.x * 64;
  const int tid  = threadIdx.x;
  const int lane = tid & 31;
  const int wave = tid >> 5;
  const int wm   = (wave & 3) * 32;    // wave row offset in tile (0..96)
  const int wn   = (wave >> 2) * 32;   // wave col offset in tile (0/32)
  const int half = lane >> 4;          // 0: lanes 0-15, 1: lanes 16-31
  const int lrow = lane & 15;

  // staging coords: A 128x32 -> 16 elems/thread (2 async b128);
  //                 B  64x32 ->  8 elems/thread (1 async b128)
  const int ar = tid >> 1,  ac = (tid & 1) * 16;
  const int br = tid >> 2,  bc = (tid & 3) * 8;

  const unsigned short* aptr = A  + (long)(row0 + ar) * Kp + ac;
  const unsigned short* bptr = BT + (long)(col0 + br) * Kp + bc;
  const unsigned lda0 = (unsigned)(size_t)&As[ar][ac];
  const unsigned lda1 = lda0 + 16;                 // next 8 bf16
  const unsigned ldb  = (unsigned)(size_t)&Bs[br][bc];

  v8f acc00 = {}, acc01 = {}, acc10 = {}, acc11 = {};

  for (int k0 = 0; k0 < Kp; k0 += 32) {
    // async global -> LDS staging (no VGPR round-trip)
    async_ld_b128(lda0, aptr + k0);
    async_ld_b128(lda1, aptr + k0 + 8);
    async_ld_b128(ldb,  bptr + k0);
    if (k0 + 32 < Kp) {
      __builtin_prefetch(aptr + k0 + 32, 0, 1);    // global_prefetch_b8
      __builtin_prefetch(bptr + k0 + 32, 0, 1);
    }
    wait_asynccnt0();      // this wave's async copies landed in LDS
    __syncthreads();       // everyone's copies visible

    // A fragments: lanes 0-15 hold K 0-15; lanes 16-31 hold K 16-31
    union { v16bf v; v8bf h[2]; } a0, a1;
    a0.h[0] = *(const v8bf*)&As[wm + lrow][half * 16];
    a0.h[1] = *(const v8bf*)&As[wm + lrow][half * 16 + 8];
    a1.h[0] = *(const v8bf*)&As[wm + 16 + lrow][half * 16];
    a1.h[1] = *(const v8bf*)&As[wm + 16 + lrow][half * 16 + 8];
    // B fragments: per-column contiguous K; lane half selects K 0-15 / 16-31
    v16bf b0 = *(const v16bf*)&Bs[wn + lrow][half * 16];
    v16bf b1 = *(const v16bf*)&Bs[wn + 16 + lrow][half * 16];

    acc00 = __builtin_amdgcn_wmma_f32_16x16x32_bf16(false, a0.v, false, b0,
                                                    (short)0, acc00, false, false);
    acc01 = __builtin_amdgcn_wmma_f32_16x16x32_bf16(false, a0.v, false, b1,
                                                    (short)0, acc01, false, false);
    acc10 = __builtin_amdgcn_wmma_f32_16x16x32_bf16(false, a1.v, false, b0,
                                                    (short)0, acc10, false, false);
    acc11 = __builtin_amdgcn_wmma_f32_16x16x32_bf16(false, a1.v, false, b1,
                                                    (short)0, acc11, false, false);
    __syncthreads();       // safe to overwrite LDS next iteration
  }

  // C layout: VGPR i -> row i + 8*half, col = lrow within each 16x16 sub-tile
#pragma unroll
  for (int mi = 0; mi < 2; ++mi) {
#pragma unroll
    for (int nj = 0; nj < 2; ++nj) {
      const v8f acc = mi ? (nj ? acc11 : acc10) : (nj ? acc01 : acc00);
      const int crow = row0 + wm + mi * 16 + half * 8;
      const int ccol = col0 + wn + nj * 16 + lrow;
      const float bv = bias ? bias[ccol] : 0.f;
#pragma unroll
      for (int i = 0; i < 8; ++i) {
        int r = crow + i;
        if (r < M) {
          float v = acc[i] + bv;
          if (relu) v = fmaxf(v, 0.f);
          C[(long)r * N + ccol] = v;
        }
      }
    }
  }
}

// ---------------- BatchNorm (training-mode batch stats) ----------------

__global__ void bn_stats(const float* __restrict__ X, int M, int N,
                         const float* __restrict__ g, const float* __restrict__ b,
                         float* __restrict__ scale, float* __restrict__ shift) {
  const int col = blockIdx.x;
  float s = 0.f, ss = 0.f;
  for (int r = threadIdx.x; r < M; r += blockDim.x) {
    float v = X[(long)r * N + col];
    s += v; ss += v * v;
  }
  __shared__ float sh1[256], sh2[256];
  sh1[threadIdx.x] = s; sh2[threadIdx.x] = ss;
  __syncthreads();
  for (int o = 128; o > 0; o >>= 1) {
    if (threadIdx.x < o) { sh1[threadIdx.x] += sh1[threadIdx.x + o];
                           sh2[threadIdx.x] += sh2[threadIdx.x + o]; }
    __syncthreads();
  }
  if (threadIdx.x == 0) {
    float m  = sh1[0] / (float)M;
    float vr = sh2[0] / (float)M - m * m;
    float sc = g[col] * rsqrtf(vr + 1e-5f);
    scale[col] = sc;
    shift[col] = b[col] - m * sc;
  }
}

__global__ void bn_apply(float* __restrict__ X, const float* __restrict__ scale,
                         const float* __restrict__ shift, long M, int N, int relu) {
  long i = (long)blockIdx.x * blockDim.x + threadIdx.x;
  if (i < M * (long)N) {
    int c = (int)(i % N);
    float v = X[i] * scale[c] + shift[c];
    if (relu) v = fmaxf(v, 0.f);
    X[i] = v;
  }
}

__global__ void col_mean(const float* __restrict__ X, int M, int N,
                         float* __restrict__ mean) {
  const int col = blockIdx.x;
  float s = 0.f;
  for (int r = threadIdx.x; r < M; r += blockDim.x) s += X[(long)r * N + col];
  __shared__ float sh[256];
  sh[threadIdx.x] = s;
  __syncthreads();
  for (int o = 128; o > 0; o >>= 1) {
    if (threadIdx.x < o) sh[threadIdx.x] += sh[threadIdx.x + o];
    __syncthreads();
  }
  if (threadIdx.x == 0) mean[col] = sh[0] / (float)M;
}

// out[j] = sum_k v[k] * W[k,j]   (self-loop edge-feature transform)
__global__ void gemv_rowvec(const float* __restrict__ v, const float* __restrict__ W,
                            float* __restrict__ out, int K, int N) {
  int j = blockIdx.x * blockDim.x + threadIdx.x;
  if (j < N) {
    float a = 0.f;
    for (int k = 0; k < K; ++k) a += v[k] * W[(long)k * N + j];
    out[j] = a;
  }
}

// ---------------- GATv2 attention (fused) ----------------
// logit[e,h] = sum_c att[h,c] * leakyrelu(xl[s]+xr[d]+ep[e], 0.2); one wave/edge

__global__ void gat_logit(const float* __restrict__ xl, const float* __restrict__ xr,
                          const float* __restrict__ ep, const float* __restrict__ ep_self,
                          const int* __restrict__ src, const int* __restrict__ dst,
                          const float* __restrict__ att, float* __restrict__ logit,
                          int E, int Nn, int H, int C) {
  const int e    = blockIdx.x * (blockDim.x >> 5) + (threadIdx.x >> 5);
  const int lane = threadIdx.x & 31;
  const int Etot = E + Nn;
  if (e >= Etot) return;
  int s, d; const float* eprow;
  if (e < E) { s = src[e]; d = dst[e]; eprow = ep + (long)e * H * C; }
  else       { s = d = e - E;          eprow = ep_self; }
  const float* xls = xl + (long)s * H * C;
  const float* xrd = xr + (long)d * H * C;
  for (int h = 0; h < H; ++h) {
    float acc = 0.f;
    for (int c = lane; c < C; c += 32) {
      float m = xls[h * C + c] + xrd[h * C + c] + eprow[h * C + c];
      m = m > 0.f ? m : 0.2f * m;
      acc += att[h * C + c] * m;
    }
    for (int o = 16; o > 0; o >>= 1) acc += __shfl_down(acc, o, 32);
    if (lane == 0) logit[(long)e * H + h] = acc;
  }
}

__global__ void seg_max_k(const float* __restrict__ logit, const int* __restrict__ dst,
                          int* __restrict__ mxord, int E, int Nn, int H) {
  long i = (long)blockIdx.x * blockDim.x + threadIdx.x;
  long total = (long)(E + Nn) * H;
  if (i >= total) return;
  int e = (int)(i / H), h = (int)(i - (long)e * H);
  int d = (e < E) ? dst[e] : (e - E);
  atomicMax(&mxord[d * H + h], f2ord(logit[i]));
}

// ex = exp(logit - mx[d]); den[d] += ex  (overwrites logit with ex)
__global__ void seg_expsum(float* __restrict__ logit, const int* __restrict__ dst,
                           const int* __restrict__ mxord, float* __restrict__ den,
                           int E, int Nn, int H) {
  long i = (long)blockIdx.x * blockDim.x + threadIdx.x;
  long total = (long)(E + Nn) * H;
  if (i >= total) return;
  int e = (int)(i / H), h = (int)(i - (long)e * H);
  int d = (e < E) ? dst[e] : (e - E);
  float ex = __expf(logit[i] - ord2f(mxord[d * H + h]));
  logit[i] = ex;
  atomicAdd(&den[d * H + h], ex);
}

// out[d,:] += (ex[e,h]/den[d,h]) * xl[s,:]
__global__ void gat_scatter(const float* __restrict__ ex, const float* __restrict__ den,
                            const float* __restrict__ xl, const int* __restrict__ src,
                            const int* __restrict__ dst, float* __restrict__ out,
                            int E, int Nn, int H, int C) {
  long i = (long)blockIdx.x * blockDim.x + threadIdx.x;
  const int HC = H * C;
  long total = (long)(E + Nn) * HC;
  if (i >= total) return;
  int e = (int)(i / HC);
  int q = (int)(i - (long)e * HC);
  int h = q / C;
  int s = (e < E) ? src[e] : (e - E);
  int d = (e < E) ? dst[e] : (e - E);
  float alpha = ex[(long)e * H + h] / den[d * H + h];
  atomicAdd(&out[(long)d * HC + q], alpha * xl[(long)s * HC + q]);
}

// ---------------- edge update: relu(xt[dst]+xt[src]+eW+b) ----------------

__global__ void edge_combine(const float* __restrict__ xt, const float* __restrict__ eW,
                             const float* __restrict__ b, const int* __restrict__ src,
                             const int* __restrict__ dst, float* __restrict__ out,
                             int E, int F) {
  long i = (long)blockIdx.x * blockDim.x + threadIdx.x;
  if (i >= (long)E * F) return;
  int e = (int)(i / F), c = (int)(i - (long)e * F);
  float v = xt[(long)dst[e] * F + c] + xt[(long)src[e] * F + c] + eW[i] + b[c];
  out[i] = fmaxf(v, 0.f);
}

// ---------------- pooling + head ----------------

__global__ void pool_scatter(const float* __restrict__ x, const int* __restrict__ batch,
                             float* __restrict__ pool, int Nn, int F) {
  long i = (long)blockIdx.x * blockDim.x + threadIdx.x;
  if (i >= (long)Nn * F) return;
  int n = (int)(i / F), c = (int)(i - (long)n * F);
  atomicAdd(&pool[batch[n] * F + c], x[i]);
}

__global__ void fc2_kernel(const float* __restrict__ h, const float* __restrict__ W,
                           const float* __restrict__ b, float* __restrict__ out,
                           int G, int F) {
  int g = blockIdx.x * blockDim.x + threadIdx.x;
  if (g < G) {
    float a = b[0];
    for (int c = 0; c < F; ++c) a += h[g * F + c] * W[c];
    out[g] = a;
  }
}

// ---------------------------------------------------------------------------
// host orchestration
// ---------------------------------------------------------------------------

extern "C" void kernel_launch(void* const* d_in, const int* in_sizes, int n_in,
                              void* d_out, int out_size, void* d_ws, size_t ws_size,
                              hipStream_t stream) {
  (void)in_sizes; (void)n_in; (void)out_size; (void)ws_size;

  // ---- input mapping (setup_inputs dict order, params flattened leaf-order) ----
  const float* x_in   = (const float*)d_in[0];   // [20000,64]
  const float* ea_in  = (const float*)d_in[1];   // [160000,16]
  const float* nmW1 = (const float*)d_in[2];  const float* nmb1 = (const float*)d_in[3];
  const float* nmW2 = (const float*)d_in[4];  const float* nmb2 = (const float*)d_in[5];
  const float* nmg  = (const float*)d_in[6];  const float* nmbe = (const float*)d_in[7];
  const float* emW1 = (const float*)d_in[8];  const float* emb1 = (const float*)d_in[9];
  const float* emW2 = (const float*)d_in[10]; const float* emb2 = (const float*)d_in[11];
  const float* emg  = (const float*)d_in[12]; const float* embe = (const float*)d_in[13];
  const float* c1Wl = (const float*)d_in[14]; const float* c1bl = (const float*)d_in[15];
  const float* c1Wr = (const float*)d_in[16]; const float* c1br = (const float*)d_in[17];
  const float* c1We = (const float*)d_in[18]; const float* c1at = (const float*)d_in[19];
  const float* c1bs = (const float*)d_in[20];
  const float* bn1g = (const float*)d_in[21]; const float* bn1b = (const float*)d_in[22];
  const float* e1Wn = (const float*)d_in[23]; const float* e1We = (const float*)d_in[24];
  const float* e1b  = (const float*)d_in[25];
  const float* eb1g = (const float*)d_in[26]; const float* eb1b = (const float*)d_in[27];
  const float* c2Wl = (const float*)d_in[28]; const float* c2bl = (const float*)d_in[29];
  const float* c2Wr = (const float*)d_in[30]; const float* c2br = (const float*)d_in[31];
  const float* c2We = (const float*)d_in[32]; const float* c2at = (const float*)d_in[33];
  const float* c2bs = (const float*)d_in[34];
  const float* bn2g = (const float*)d_in[35]; const float* bn2b = (const float*)d_in[36];
  // eu2/ebn2 (d_in[37..41]) dead wrt output -> skipped
  const float* f1W  = (const float*)d_in[42]; const float* f1b = (const float*)d_in[43];
  const float* f2W  = (const float*)d_in[44]; const float* f2b = (const float*)d_in[45];
  const int*   eidx = (const int*)d_in[46];   // [2, E]
  const int*   batch= (const int*)d_in[47];   // [N]
  const int* src = eidx;
  const int* dst = eidx + N_EDGES;

  const int Nn = N_NODES, E = N_EDGES, Etot = N_EDGES + N_NODES;
  const long NnP = 20096;   // ceil128(20000): A-buffer row padding for 128-row tiles

  // ---- workspace bump allocator ----
  char* wp = (char*)d_ws;
  auto alloc = [&](size_t bytes) -> void* {
    void* r = (void*)wp;
    wp += (bytes + 255) & ~(size_t)255;
    return r;
  };
  unsigned short* bfA = (unsigned short*)alloc((size_t)NnP * 1024 * 2);    // node A operands
  unsigned short* bfA2= (unsigned short*)alloc((size_t)E * 128 * 2);       // edge A operands
  unsigned short* bfB = (unsigned short*)alloc((size_t)1024 * 1024 * 2);   // B^T
  float* h512   = (float*)alloc((size_t)Nn * 512 * 4);
  float* x256   = (float*)alloc((size_t)Nn * 256 * 4);
  float* e64    = (float*)alloc((size_t)E * 64 * 4);
  float* etmp   = (float*)alloc((size_t)E * 64 * 4);
  float* xl1    = (float*)alloc((size_t)Nn * 1024 * 4);
  float* xr1    = (float*)alloc((size_t)Nn * 1024 * 4);
  float* ep1    = (float*)alloc((size_t)E * 1024 * 4);
  float* epself = (float*)alloc((size_t)1024 * 4);
  float* cmean  = (float*)alloc((size_t)128 * 4);
  float* logit  = (float*)alloc((size_t)Etot * 4 * 4);
  int*   mxord  = (int*)  alloc((size_t)Nn * 4 * 4);
  float* den    = (float*)alloc((size_t)Nn * 4 * 4);
  float* xB     = (float*)alloc((size_t)Nn * 1024 * 4);
  float* bnsc   = (float*)alloc((size_t)1024 * 4);
  float* bnsh   = (float*)alloc((size_t)1024 * 4);
  float* xt1    = (float*)alloc((size_t)Nn * 128 * 4);
  float* eW1    = (float*)alloc((size_t)E * 128 * 4);
  float* e128   = (float*)alloc((size_t)E * 128 * 4);
  float* xl2    = (float*)alloc((size_t)Nn * 256 * 4);
  float* xr2    = (float*)alloc((size_t)Nn * 256 * 4);
  float* ep2    = (float*)alloc((size_t)E * 256 * 4);
  float* xC     = (float*)alloc((size_t)Nn * 256 * 4);
  float* pool   = (float*)alloc((size_t)N_GRAPHS * 256 * 4);
  float* hfc    = (float*)alloc((size_t)N_GRAPHS * 64 * 4);

  auto gblk = [](long n) { return (unsigned)((n + 255) / 256); };
  auto kpad = [](int K) { return (K + 31) & ~31; };
  auto cvtA = [&](const float* s, unsigned short* d, long M, int K) {
    int Kp = kpad(K);
    cvt_pad_bf16<<<gblk(M * Kp), 256, 0, stream>>>(s, d, M, K, Kp);
  };
  auto cvtB = [&](const float* s, unsigned short* d, int K, int N) {
    int Kp = kpad(K);
    cvtT_bf16<<<gblk((long)N * Kp), 256, 0, stream>>>(s, d, K, N, Kp);
  };
  auto gemm = [&](const unsigned short* A, const unsigned short* BT, const float* bias,
                  float* C, int M, int N, int K, int relu) {
    dim3 grid(N / 64, (M + 127) / 128);
    gemm_bf16_wmma<<<grid, 256, 0, stream>>>(A, BT, bias, C, M, N, kpad(K), relu);
  };
  auto bn = [&](float* X, long M, int N, const float* g, const float* b, int relu) {
    bn_stats<<<N, 256, 0, stream>>>(X, (int)M, N, g, b, bnsc, bnsh);
    bn_apply<<<gblk(M * N), 256, 0, stream>>>(X, bnsc, bnsh, M, N, relu);
  };
  const int NEG_INF_ORD = (int)0x807FFFFFu;  // f2ord(-inf)

  // ===== node MLP =====
  cvtA(x_in, bfA, Nn, 64);
  cvtB(nmW1, bfB, 64, 512);
  gemm(bfA, bfB, nmb1, h512, Nn, 512, 64, 1);
  cvtA(h512, bfA, Nn, 512);
  cvtB(nmW2, bfB, 512, 256);
  gemm(bfA, bfB, nmb2, x256, Nn, 256, 512, 0);
  bn(x256, Nn, 256, nmg, nmbe, 1);

  // ===== edge MLP =====
  cvtA(ea_in, bfA2, E, 16);                 // K=16 -> Kp=32 zero-padded
  cvtB(emW1, bfB, 16, 64);
  gemm(bfA2, bfB, emb1, etmp, E, 64, 16, 1);
  cvtA(etmp, bfA2, E, 64);
  cvtB(emW2, bfB, 64, 64);
  gemm(bfA2, bfB, emb2, e64, E, 64, 64, 0);
  bn(e64, E, 64, emg, embe, 1);

  // ===== conv1: GATv2 H=4 C=256 =====
  cvtA(x256, bfA, Nn, 256);
  cvtB(c1Wl, bfB, 256, 1024);
  gemm(bfA, bfB, c1bl, xl1, Nn, 1024, 256, 0);
  cvtB(c1Wr, bfB, 256, 1024);
  gemm(bfA, bfB, c1br, xr1, Nn, 1024, 256, 0);
  cvtA(e64, bfA2, E, 64);
  cvtB(c1We, bfB, 64, 1024);
  gemm(bfA2, bfB, nullptr, ep1, E, 1024, 64, 0);
  col_mean<<<64, 256, 0, stream>>>(e64, E, 64, cmean);
  gemv_rowvec<<<(1024 + 255) / 256, 256, 0, stream>>>(cmean, c1We, epself, 64, 1024);

  gat_logit<<<(Etot + 3) / 4, 128, 0, stream>>>(xl1, xr1, ep1, epself, src, dst,
                                                c1at, logit, E, Nn, 4, 256);
  fill_i32<<<gblk((long)Nn * 4), 256, 0, stream>>>(mxord, NEG_INF_ORD, (long)Nn * 4);
  fill_f32<<<gblk((long)Nn * 4), 256, 0, stream>>>(den, 0.f, (long)Nn * 4);
  fill_f32<<<gblk((long)Nn * 1024), 256, 0, stream>>>(xB, 0.f, (long)Nn * 1024);
  seg_max_k<<<gblk((long)Etot * 4), 256, 0, stream>>>(logit, dst, mxord, E, Nn, 4);
  seg_expsum<<<gblk((long)Etot * 4), 256, 0, stream>>>(logit, dst, mxord, den, E, Nn, 4);
  gat_scatter<<<gblk((long)Etot * 1024), 256, 0, stream>>>(logit, den, xl1, src, dst,
                                                           xB, E, Nn, 4, 256);
  bias_relu_inplace<<<gblk((long)Nn * 1024), 256, 0, stream>>>(xB, c1bs, Nn, 1024);
  bn(xB, Nn, 1024, bn1g, bn1b, 0);

  // ===== eu1: e = bn(relu(xt[dst]+xt[src]+e@We+b)) =====
  cvtA(xB, bfA, Nn, 1024);                  // reused for conv2 xl/xr below
  cvtB(e1Wn, bfB, 1024, 128);
  gemm(bfA, bfB, nullptr, xt1, Nn, 128, 1024, 0);
  cvtB(e1We, bfB, 64, 128);
  gemm(bfA2, bfB, nullptr, eW1, E, 128, 64, 0);   // bfA2 still bf16(e64)
  edge_combine<<<gblk((long)E * 128), 256, 0, stream>>>(xt1, eW1, e1b, src, dst,
                                                        e128, E, 128);
  bn(e128, E, 128, eb1g, eb1b, 0);

  // ===== conv2: GATv2 H=4 C=64 =====
  cvtB(c2Wl, bfB, 1024, 256);
  gemm(bfA, bfB, c2bl, xl2, Nn, 256, 1024, 0);
  cvtB(c2Wr, bfB, 1024, 256);
  gemm(bfA, bfB, c2br, xr2, Nn, 256, 1024, 0);
  cvtA(e128, bfA2, E, 128);
  cvtB(c2We, bfB, 128, 256);
  gemm(bfA2, bfB, nullptr, ep2, E, 256, 128, 0);
  col_mean<<<128, 256, 0, stream>>>(e128, E, 128, cmean);
  gemv_rowvec<<<(256 + 255) / 256, 256, 0, stream>>>(cmean, c2We, epself, 128, 256);

  gat_logit<<<(Etot + 3) / 4, 128, 0, stream>>>(xl2, xr2, ep2, epself, src, dst,
                                                c2at, logit, E, Nn, 4, 64);
  fill_i32<<<gblk((long)Nn * 4), 256, 0, stream>>>(mxord, NEG_INF_ORD, (long)Nn * 4);
  fill_f32<<<gblk((long)Nn * 4), 256, 0, stream>>>(den, 0.f, (long)Nn * 4);
  fill_f32<<<gblk((long)Nn * 256), 256, 0, stream>>>(xC, 0.f, (long)Nn * 256);
  seg_max_k<<<gblk((long)Etot * 4), 256, 0, stream>>>(logit, dst, mxord, E, Nn, 4);
  seg_expsum<<<gblk((long)Etot * 4), 256, 0, stream>>>(logit, dst, mxord, den, E, Nn, 4);
  gat_scatter<<<gblk((long)Etot * 256), 256, 0, stream>>>(logit, den, xl2, src, dst,
                                                          xC, E, Nn, 4, 64);
  bias_relu_inplace<<<gblk((long)Nn * 256), 256, 0, stream>>>(xC, c2bs, Nn, 256);
  bn(xC, Nn, 256, bn2g, bn2b, 0);

  // ===== global_add_pool + head =====
  fill_f32<<<gblk((long)N_GRAPHS * 256), 256, 0, stream>>>(pool, 0.f,
                                                           (long)N_GRAPHS * 256);
  pool_scatter<<<gblk((long)Nn * 256), 256, 0, stream>>>(xC, batch, pool, Nn, 256);
  cvtA(pool, bfA, N_GRAPHS, 256);
  cvtB(f1W, bfB, 256, 64);
  gemm(bfA, bfB, f1b, hfc, N_GRAPHS, 64, 256, 1);
  fc2_kernel<<<1, 64, 0, stream>>>(hfc, f2W, f2b, (float*)d_out, N_GRAPHS, 64);
}
